// PermutationModule_43808666419472
// MI455X (gfx1250) — matrix operand order repaired and checked by reading, not compiled, not run on hardware
//
#include <hip/hip_runtime.h>
#include <hip/hip_bf16.h>

// Problem constants (match reference)
#define BB 4
#define NN 512
#define DD 512
#define PP 128
#define MTOT (BB * NN)   // 2048 rows
#define KTILES (DD / 32) // 16 k-steps of 32

typedef __attribute__((ext_vector_type(16))) _Float16 v16h;
typedef __attribute__((ext_vector_type(8)))  _Float16 v8h;
typedef __attribute__((ext_vector_type(8)))  float    v8f;

typedef __attribute__((ext_vector_type(4))) unsigned int tdm_u32x4;
typedef __attribute__((ext_vector_type(8))) int          tdm_i32x8;
typedef __attribute__((ext_vector_type(4))) int          tdm_i32x4;

#if defined(__gfx1250__) && __has_builtin(__builtin_amdgcn_tensor_load_to_lds)
#define HAVE_TDM 1
#else
#define HAVE_TDM 0
#endif

// ---------------------------------------------------------------------------
// f32 -> f16 conversion (elementwise) for the activation matrix
// ---------------------------------------------------------------------------
__global__ __launch_bounds__(256) void cvt_f32_to_f16(const float* __restrict__ src,
                                                      _Float16* __restrict__ dst,
                                                      int n) {
    int i = blockIdx.x * 256 + threadIdx.x;
    if (i < n) dst[i] = (_Float16)src[i];
}

// ---------------------------------------------------------------------------
// Pack W [K=512, N=128] f32 row-major into WMMA B-fragment order (f16):
// fragment id t = (kb*8 + ntile)*32 + lane, 16 contiguous halves per lane:
//   n = ntile*16 + (lane&15),  k = kb*32 + (lane>>4)*16 + i
// ---------------------------------------------------------------------------
__global__ __launch_bounds__(256) void pack_w_f16(const float* __restrict__ W,
                                                  _Float16* __restrict__ Pk) {
    int t = blockIdx.x * 256 + threadIdx.x;      // 16*8*32 = 4096 fragment-lanes
    if (t >= KTILES * 8 * 32) return;
    int lane  = t & 31;
    int ntile = (t >> 5) & 7;
    int kb    = t >> 8;
    int n     = ntile * 16 + (lane & 15);
    int kbase = kb * 32 + (lane >> 4) * 16;
    _Float16* dst = Pk + (size_t)t * 16;
#pragma unroll
    for (int i = 0; i < 16; ++i)
        dst[i] = (_Float16)W[(size_t)(kbase + i) * PP + n];
}

// ---------------------------------------------------------------------------
// TDM: DMA a 2D tile (rows x 4KB, 8KB row stride) from global into LDS.
// D# built per ISA 8.3/8.4: data_size=8B units, tile_dim0=512 (=4KB),
// tensor_dim0 = tensor_dim0_stride = 1024 (=8KB), tile_dim1 = rows.
// ---------------------------------------------------------------------------
#if HAVE_TDM
__device__ __forceinline__ void tdm_load_strip(unsigned long long gaddr,
                                               unsigned int lds_addr,
                                               unsigned int rows) {
    tdm_u32x4 g0;
    g0[0] = 1u;                                           // count=1, user mode
    g0[1] = lds_addr;                                     // LDS byte address
    g0[2] = (unsigned int)(gaddr & 0xFFFFFFFFull);        // global_addr[31:0]
    g0[3] = (unsigned int)((gaddr >> 32) & 0x01FFFFFFull) // global_addr[56:32]
          | (2u << 30);                                   // type = 2 (image)

    tdm_i32x8 g1;
    g1[0] = (int)(3u << 16);                  // data_size = 3 (8B); mask=0
    g1[1] = (int)((1024u & 0xFFFFu) << 16);   // tensor_dim0[15:0]=1024
    g1[2] = (int)((1024u >> 16) | ((rows & 0xFFFFu) << 16));  // dim0 hi | tensor_dim1
    g1[3] = (int)((rows >> 16) | (512u << 16));               // dim1 hi | tile_dim0=512
    g1[4] = (int)(rows & 0xFFFFu);            // tile_dim1=rows, tile_dim2=0
    g1[5] = (int)1024;                        // tensor_dim0_stride[31:0] (8B units)
    g1[6] = 0;                                // stride hi | dim1_stride lo
    g1[7] = 0;

    tdm_i32x4 z4 = {0, 0, 0, 0};
#if __clang_major__ >= 23
    tdm_i32x8 z8 = {0, 0, 0, 0, 0, 0, 0, 0};
    __builtin_amdgcn_tensor_load_to_lds(g0, g1, z4, z4, z8, 0);
#else
    __builtin_amdgcn_tensor_load_to_lds(g0, g1, z4, z4, 0);
#endif
}
#endif

// ---------------------------------------------------------------------------
// WMMA GEMM: C[M,128] = A[M,512](f16) * W(packed f16) (+bias) (opt relu)
// Block = 256 threads = 8 waves. Wave computes a 16x64 strip (4 n-tiles).
// B strip staged via TDM into LDS in two 32KB phases (8 k-steps each);
// all 8 waves then feed WMMA from ds_load_b128. A streams from global.
// grid = (M/128, 2): y selects n-columns [0,64) or [64,128).
// ---------------------------------------------------------------------------
__global__ __launch_bounds__(256) void gemm_wmma_f16(const _Float16* __restrict__ A,
                                                     const _Float16* __restrict__ Wp,
                                                     const float* __restrict__ bias,
                                                     float* __restrict__ C,
                                                     int hasBias, int doRelu) {
    __shared__ _Float16 bsh[8 * 2048];            // 8 k-chunks * 4KB = 32KB

    const int lane  = threadIdx.x & 31;
    const int wave  = threadIdx.x >> 5;
    const int l15   = lane & 15;
    const int h16   = lane >> 4;                  // 0 or 1 (half-wave)
    const int tileM = blockIdx.x * 128 + wave * 16;
    const int nbase = blockIdx.y * 4;             // first n-tile index (of 8)

    v8f acc0 = {}, acc1 = {}, acc2 = {}, acc3 = {};

    for (int ph = 0; ph < 2; ++ph) {
        if (ph) __syncthreads();                  // protect LDS reuse
#if HAVE_TDM
        if (wave == 0) {
            unsigned long long gaddr = (unsigned long long)(uintptr_t)Wp
                                     + (unsigned long long)nbase * 1024ull
                                     + (unsigned long long)ph * 65536ull;
            unsigned int laddr = (unsigned int)(uintptr_t)&bsh[0];
            tdm_load_strip(gaddr, laddr, 8u);
            __builtin_amdgcn_s_wait_tensorcnt(0);
        }
#else
        {   // cooperative fallback copy: 8 chunks of 4KB, src stride 8KB
            const char* src = (const char*)Wp + (size_t)nbase * 1024 + (size_t)ph * 65536;
            for (int u = threadIdx.x; u < 2048; u += 256) {   // 16B units
                int chunk = u >> 8, within = u & 255;
                *(uint4*)((char*)bsh + u * 16) =
                    *(const uint4*)(src + (size_t)chunk * 8192 + (size_t)within * 16);
            }
        }
#endif
        __syncthreads();

#pragma unroll
        for (int kk = 0; kk < 8; ++kk) {
            const int kb = ph * 8 + kk;
            // A fragment (ISA 7.12.2, 16-bit A 16x32): two contiguous b128 loads.
            const _Float16* arow = A + (size_t)(tileM + l15) * DD + kb * 32 + h16 * 16;
            v8h a_lo = *(const v8h*)(arow);
            v8h a_hi = *(const v8h*)(arow + 8);
            v16h af;
#pragma unroll
            for (int i = 0; i < 8; ++i) { af[i] = a_lo[i]; af[i + 8] = a_hi[i]; }

            // B fragments from LDS: 32 contiguous bytes per lane.
            const _Float16* bp = &bsh[kk * 2048 + lane * 16];
            v16h b0 = *(const v16h*)(bp);
            v16h b1 = *(const v16h*)(bp + 512);
            v16h b2 = *(const v16h*)(bp + 1024);
            v16h b3 = *(const v16h*)(bp + 1536);

            acc0 = __builtin_amdgcn_wmma_f32_16x16x32_f16(false, af, false, b0,
                                                          (short)0, acc0, false, false);
            acc1 = __builtin_amdgcn_wmma_f32_16x16x32_f16(false, af, false, b1,
                                                          (short)0, acc1, false, false);
            acc2 = __builtin_amdgcn_wmma_f32_16x16x32_f16(false, af, false, b2,
                                                          (short)0, acc2, false, false);
            acc3 = __builtin_amdgcn_wmma_f32_16x16x32_f16(false, af, false, b3,
                                                          (short)0, acc3, false, false);
        }
    }

    // C/D layout: VGPR r -> M = (lane>=16 ? 8 : 0) + r, N = lane&15
    const int colBase = nbase * 16 + l15;
    float bv0 = 0.f, bv1 = 0.f, bv2 = 0.f, bv3 = 0.f;
    if (hasBias) {
        bv0 = bias[colBase];
        bv1 = bias[colBase + 16];
        bv2 = bias[colBase + 32];
        bv3 = bias[colBase + 48];
    }
#pragma unroll
    for (int r = 0; r < 8; ++r) {
        float* crow = C + (size_t)(tileM + h16 * 8 + r) * PP + colBase;
        float v0 = acc0[r] + bv0;
        float v1 = acc1[r] + bv1;
        float v2 = acc2[r] + bv2;
        float v3 = acc3[r] + bv3;
        if (doRelu) {
            v0 = fmaxf(v0, 0.f); v1 = fmaxf(v1, 0.f);
            v2 = fmaxf(v2, 0.f); v3 = fmaxf(v3, 0.f);
        }
        crow[0]  = v0;
        crow[16] = v1;
        crow[32] = v2;
        crow[48] = v3;
    }
}

// ---------------------------------------------------------------------------
// start/end heads: out[r] = dot(H[r,:], W2) + b2   (P=128, trivially small)
// ---------------------------------------------------------------------------
__global__ __launch_bounds__(256) void head_kernel(const float* __restrict__ sH,
                                                   const float* __restrict__ eH,
                                                   const float* __restrict__ Ws2,
                                                   const float* __restrict__ bs2,
                                                   const float* __restrict__ We2,
                                                   const float* __restrict__ be2,
                                                   float* __restrict__ outStart,
                                                   float* __restrict__ outEnd) {
    int r = blockIdx.x * 256 + threadIdx.x;
    if (r >= MTOT) return;
    const float* srow = sH + (size_t)r * PP;
    const float* erow = eH + (size_t)r * PP;
    float as = 0.f, ae = 0.f;
#pragma unroll 4
    for (int p = 0; p < PP; ++p) {
        as = fmaf(srow[p], Ws2[p], as);
        ae = fmaf(erow[p], We2[p], ae);
    }
    outStart[r] = as + bs2[0];
    outEnd[r]   = ae + be2[0];
}

// ---------------------------------------------------------------------------
// bigram: out[b,i,j] = bo + sum_p relu(left[b,j,p] + right[b,i,p]) * Wo[p]
// grid = (N/256, N, B); block = 256 (thread owns one j).
// right[b,i,:] and Wo staged in LDS; left rows are L2-resident (1 MB vs 192 MB).
// ---------------------------------------------------------------------------
__global__ __launch_bounds__(256) void bigram_kernel(const float* __restrict__ left,
                                                     const float* __restrict__ right,
                                                     const float* __restrict__ Wo,
                                                     const float* __restrict__ bo,
                                                     float* __restrict__ out) {
    __shared__ float rS[PP];
    __shared__ float wS[PP];
    const int j = blockIdx.x * 256 + threadIdx.x;
    const int i = blockIdx.y;
    const int b = blockIdx.z;

    const float4* lrow = (const float4*)(left + ((size_t)b * NN + j) * PP);
    __builtin_prefetch(lrow, 0, 0);               // global_prefetch_b8
    __builtin_prefetch(lrow + 16, 0, 0);

    const float* rrow = right + ((size_t)b * NN + i) * PP;
    if (threadIdx.x < PP)          rS[threadIdx.x]      = rrow[threadIdx.x];
    else if (threadIdx.x < 2 * PP) wS[threadIdx.x - PP] = Wo[threadIdx.x - PP];
    __syncthreads();

    float acc = 0.f;
#pragma unroll 8
    for (int p4 = 0; p4 < PP / 4; ++p4) {
        float4 l4 = lrow[p4];
        int p = p4 * 4;
        acc = fmaf(fmaxf(l4.x + rS[p + 0], 0.f), wS[p + 0], acc);
        acc = fmaf(fmaxf(l4.y + rS[p + 1], 0.f), wS[p + 1], acc);
        acc = fmaf(fmaxf(l4.z + rS[p + 2], 0.f), wS[p + 2], acc);
        acc = fmaf(fmaxf(l4.w + rS[p + 3], 0.f), wS[p + 3], acc);
    }
    out[((size_t)b * NN + i) * NN + j] = acc + bo[0];
}

// ---------------------------------------------------------------------------
// Launch
// ---------------------------------------------------------------------------
extern "C" void kernel_launch(void* const* d_in, const int* in_sizes, int n_in,
                              void* d_out, int out_size, void* d_ws, size_t ws_size,
                              hipStream_t stream) {
    const float* input = (const float*)d_in[0];
    const float* Wl    = (const float*)d_in[1];
    const float* bl    = (const float*)d_in[2];
    const float* Wr    = (const float*)d_in[3];
    const float* Wo    = (const float*)d_in[4];
    const float* bo    = (const float*)d_in[5];
    const float* Ws1   = (const float*)d_in[6];
    const float* bs1   = (const float*)d_in[7];
    const float* Ws2   = (const float*)d_in[8];
    const float* bs2   = (const float*)d_in[9];
    const float* We1   = (const float*)d_in[10];
    const float* be1   = (const float*)d_in[11];
    const float* We2   = (const float*)d_in[12];
    const float* be2   = (const float*)d_in[13];

    // Workspace layout (all offsets 16B aligned)
    char* ws = (char*)d_ws;
    _Float16* inputH = (_Float16*)(ws);                               // 2048*512*2 = 2 MB
    _Float16* WlP    = (_Float16*)(ws + 2097152);                     // packed, 128 KB each
    _Float16* WrP    = (_Float16*)(ws + 2097152 + 131072);
    _Float16* Ws1P   = (_Float16*)(ws + 2097152 + 2 * 131072);
    _Float16* We1P   = (_Float16*)(ws + 2097152 + 3 * 131072);
    float*    left   = (float*)(ws + 2621440);                        // 1 MB each
    float*    right  = (float*)(ws + 2621440 + 1048576);
    float*    sH     = (float*)(ws + 2621440 + 2 * 1048576);
    float*    eH     = (float*)(ws + 2621440 + 3 * 1048576);

    // 1) convert activations; pack+convert weights into B-fragment order
    {
        int n = MTOT * DD;
        cvt_f32_to_f16<<<(n + 255) / 256, 256, 0, stream>>>(input, inputH, n);
        int nf = KTILES * 8 * 32;   // 4096 fragment-lanes per matrix
        pack_w_f16<<<(nf + 255) / 256, 256, 0, stream>>>(Wl,  WlP);
        pack_w_f16<<<(nf + 255) / 256, 256, 0, stream>>>(Wr,  WrP);
        pack_w_f16<<<(nf + 255) / 256, 256, 0, stream>>>(Ws1, Ws1P);
        pack_w_f16<<<(nf + 255) / 256, 256, 0, stream>>>(We1, We1P);
    }

    // 2) four WMMA GEMMs (TDM-staged B in LDS; 4 WMMAs per k-step per wave)
    dim3 ggrid(MTOT / 128, 2);
    gemm_wmma_f16<<<ggrid, 256, 0, stream>>>(inputH, WlP,  bl,  left,  1, 0);
    gemm_wmma_f16<<<ggrid, 256, 0, stream>>>(inputH, WrP,  bl,  right, 0, 0);
    gemm_wmma_f16<<<ggrid, 256, 0, stream>>>(inputH, Ws1P, bs1, sH,    1, 1);
    gemm_wmma_f16<<<ggrid, 256, 0, stream>>>(inputH, We1P, be1, eH,    1, 1);

    // 3) start/end heads
    float* outBigram = (float*)d_out;                 // B*N*N = 1,048,576 floats
    float* outStart  = outBigram + (size_t)BB * NN * NN;
    float* outEnd    = outStart + (size_t)BB * NN;
    head_kernel<<<MTOT / 256, 256, 0, stream>>>(sH, eH, Ws2, bs2, We2, be2,
                                                outStart, outEnd);

    // 4) bigram
    dim3 bgrid(NN / 256, NN, BB);                     // (2, 512, 4)
    bigram_kernel<<<bgrid, 256, 0, stream>>>(left, right, Wo, bo, outBigram);
}
